// JambaLayer_49589692400336
// MI455X (gfx1250) — matrix operand order repaired and checked by reading, not compiled, not run on hardware
//
#include <hip/hip_runtime.h>

// ---------------------------------------------------------------------------
// Jamba layer (Mamba + SwiGLU) for MI455X / gfx1250.
// GEMMs: v_wmma_f32_16x16x32_bf16, async global->LDS b128 staging, fp32 accum.
// ---------------------------------------------------------------------------

#define BDIM 256
#define LDSU 20   // LDS row stride in uints (80 B) -> 16B-aligned b128 ops

typedef __attribute__((ext_vector_type(16))) __bf16 v16bf;
typedef __attribute__((ext_vector_type(8)))  float  v8f;
typedef int v4i __attribute__((vector_size(16)));   // matches builtin param type

static constexpr int Bc  = 2;
static constexpr int Lc  = 2048;
static constexpr int DM  = 1024;   // d_model
static constexpr int DI  = 2048;   // d_inner
static constexpr int DTR = 64;     // dt_rank
static constexpr int DSt = 16;     // d_state
static constexpr int DFF = 4096;   // SwiGLU hidden
static constexpr int Mtok = Bc * Lc;   // 4096 tokens

// ---- gfx1250 async global->LDS path (builtin confirmed present) -------------
#if defined(__has_builtin)
#  if __has_builtin(__builtin_amdgcn_global_load_async_to_lds_b128)
#    define HAVE_ASYNC_LDS 1
#  endif
#endif
#ifndef HAVE_ASYNC_LDS
#  define HAVE_ASYNC_LDS 0
#endif

#if HAVE_ASYNC_LDS
typedef __attribute__((address_space(1))) v4i* gv4i_p;   // global int4*
typedef __attribute__((address_space(3))) v4i* lv4i_p;   // LDS int4*
__device__ __forceinline__ void async_ld_b128(const unsigned int* g, unsigned int* l) {
  // AS1 address == generic address for global; AS3 offset == low 32 bits of flat.
  __builtin_amdgcn_global_load_async_to_lds_b128(
      (gv4i_p)(unsigned long long)g,
      (lv4i_p)(unsigned int)(unsigned long long)l,
      0, 0);
}
#endif

__device__ __forceinline__ void wait_async0() {
#if defined(__has_builtin) && __has_builtin(__builtin_amdgcn_s_wait_asynccnt)
  __builtin_amdgcn_s_wait_asynccnt(0);
#else
  asm volatile("s_wait_asynccnt 0x0" ::: "memory");
#endif
}

__device__ __forceinline__ unsigned short f2bf(float f) {
  unsigned int u = __float_as_uint(f);
  u += 0x7FFFu + ((u >> 16) & 1u);          // round-to-nearest-even
  return (unsigned short)(u >> 16);
}
__device__ __forceinline__ float silu_f(float x) { return x / (1.f + __expf(-x)); }

union FragU { v16bf v; unsigned int u[8]; uint4 q[2]; };

// A-matrix 16x32 bf16 fragment (ISA 7.12.2). Per-lane uints form two
// contiguous runs: p0[4h..4h+3] and p0[8+4h..8+4h+3]  -> two ds_load_b128.
__device__ __forceinline__ v16bf frag_a(const unsigned int* s, int rowBase, int lane) {
  const unsigned int* p = s + (size_t)(rowBase + (lane & 15)) * LDSU + ((lane >> 4) & 1) * 4;
  FragU f;
  f.q[0] = *(const uint4*)p;
  f.q[1] = *(const uint4*)(p + 8);
  return f.v;
}

// B-matrix 32x16 bf16 fragment: lane n=l%16 holds column n; same K order as A:
// uints p0[4h..4h+3] and p0[8+4h..8+4h+3].
__device__ __forceinline__ v16bf frag_b(const unsigned int* s, int rowBase, int lane) {
  const unsigned int* p = s + (size_t)(rowBase + (lane & 15)) * LDSU + ((lane >> 4) & 1) * 4;
  FragU f;
  f.q[0] = *(const uint4*)p;
  f.q[1] = *(const uint4*)(p + 8);
  return f.v;
}

__device__ __forceinline__ v8f zero8() {
  v8f z = {0.f, 0.f, 0.f, 0.f, 0.f, 0.f, 0.f, 0.f};
  return z;
}

// Stage one ROWSx32 bf16 (ROWSx16 uint) tile into LDS with b128 ops.
template <int ROWS>
__device__ __forceinline__ void stage_tile(const unsigned int* __restrict__ G,
                                           unsigned int* __restrict__ S,
                                           int row0, int Ku, int kc, int tid) {
#pragma unroll
  for (int i = tid; i < ROWS * 4; i += BDIM) {
    const int r = i >> 2, c = (i & 3) * 4;
    const unsigned int* gp = G + (size_t)(row0 + r) * Ku + kc + c;
    unsigned int* lp = S + r * LDSU + c;
#if HAVE_ASYNC_LDS
    async_ld_b128(gp, lp);
#else
    *(uint4*)lp = *(const uint4*)gp;
#endif
  }
}

// Guarded variant (rows >= N are zero-filled).
template <int ROWS>
__device__ __forceinline__ void stage_tile_guard(const unsigned int* __restrict__ G,
                                                 unsigned int* __restrict__ S,
                                                 int row0, int Ku, int kc, int N, int tid) {
#pragma unroll
  for (int i = tid; i < ROWS * 4; i += BDIM) {
    const int r = i >> 2, c = (i & 3) * 4;
    const int n = row0 + r;
    unsigned int* lp = S + r * LDSU + c;
    if (n < N) {
      const unsigned int* gp = G + (size_t)n * Ku + kc + c;
#if HAVE_ASYNC_LDS
      async_ld_b128(gp, lp);
#else
      *(uint4*)lp = *(const uint4*)gp;
#endif
    } else {
      uint4 z; z.x = z.y = z.z = z.w = 0u;
      *(uint4*)lp = z;
    }
  }
}

__device__ __forceinline__ void tiles_ready() {
#if HAVE_ASYNC_LDS
  wait_async0();
#endif
  __syncthreads();
}

// ---------------------------------------------------------------------------
// GEMM: C[M,N] = epi( A[M,K] * W[N,K]^T )   (A, W in bf16; C fp32, ldc = N)
// EPI 0: none | 1: softplus(acc + bias[n]) | 2: acc + resid[m,n]
// Block tile 128x128, 8 waves (4 in M x 2 in N), wave tile 32x64.
// ---------------------------------------------------------------------------
template <int EPI>
__global__ __launch_bounds__(BDIM) void gemm_nt(
    const unsigned short* __restrict__ A, const unsigned short* __restrict__ W,
    float* __restrict__ C, const float* __restrict__ bias,
    const float* __restrict__ resid, int M, int N, int K)
{
  __shared__ unsigned int sA[128 * LDSU];
  __shared__ unsigned int sB[128 * LDSU];

  const int tid  = threadIdx.x;
  const int lane = tid & 31;
  const int wave = tid >> 5;
  const int wm   = (wave & 3) * 32;   // wave M offset in block
  const int wn   = (wave >> 2) * 64;  // wave N offset in block
  const int m0   = blockIdx.y * 128;
  const int n0   = blockIdx.x * 128;

  const unsigned int* Ag = (const unsigned int*)A;
  const unsigned int* Wg = (const unsigned int*)W;
  const int Ku = K >> 1;

  v8f acc[2][4];
#pragma unroll
  for (int t = 0; t < 2; ++t)
#pragma unroll
    for (int j = 0; j < 4; ++j) acc[t][j] = zero8();

  for (int k0 = 0; k0 < K; k0 += 32) {
    const int kc = k0 >> 1;
    __syncthreads();
    stage_tile<128>(Ag, sA, m0, Ku, kc, tid);
    stage_tile_guard<128>(Wg, sB, n0, Ku, kc, N, tid);
    if (k0 + 32 < K) {  // hint next A tile toward cache (global_prefetch_b8)
      __builtin_prefetch(&Ag[(size_t)(m0 + (tid & 127)) * Ku + kc + 16], 0, 1);
    }
    tiles_ready();

    v16bf af[2], bfv[4];
#pragma unroll
    for (int t = 0; t < 2; ++t) af[t] = frag_a(sA, wm + t * 16, lane);
#pragma unroll
    for (int j = 0; j < 4; ++j) bfv[j] = frag_b(sB, wn + j * 16, lane);
#pragma unroll
    for (int t = 0; t < 2; ++t)
#pragma unroll
      for (int j = 0; j < 4; ++j)
        acc[t][j] = __builtin_amdgcn_wmma_f32_16x16x32_bf16(
            false, af[t], false, bfv[j], (short)0, acc[t][j], false, false);
  }

  // C/D layout: lanes 0-15 col = lane, row r; lanes 16-31 row 8+r.
  const int colL  = lane & 15;
  const int rhalf = (lane >> 4) * 8;
#pragma unroll
  for (int t = 0; t < 2; ++t) {
#pragma unroll
    for (int j = 0; j < 4; ++j) {
      const int col = n0 + wn + j * 16 + colL;
      if (col >= N) continue;
      const int rowb = m0 + wm + t * 16 + rhalf;
#pragma unroll
      for (int r = 0; r < 8; ++r) {
        float v = acc[t][j][r];
        const size_t off = (size_t)(rowb + r) * N + col;
        if (EPI == 1) { v += bias[col]; v = (v > 20.f) ? v : log1pf(__expf(v)); }
        else if (EPI == 2) { v += resid[off]; }
        C[off] = v;
      }
    }
  }
}

// ---------------------------------------------------------------------------
// Fused SwiGLU dual GEMM: O_bf16[M,N] = bf16( silu(A*W1^T) * (A*W2^T) )
// Block tile 128x64, 8 waves (4 in M x 2 in N), wave tile 32x32, two acc sets.
// ---------------------------------------------------------------------------
__global__ __launch_bounds__(BDIM) void gemm_dual_swiglu(
    const unsigned short* __restrict__ A, const unsigned short* __restrict__ W1,
    const unsigned short* __restrict__ W2, unsigned short* __restrict__ O,
    int M, int N, int K)
{
  __shared__ unsigned int sA[128 * LDSU];
  __shared__ unsigned int sB1[64 * LDSU];
  __shared__ unsigned int sB2[64 * LDSU];

  const int tid  = threadIdx.x;
  const int lane = tid & 31;
  const int wave = tid >> 5;
  const int wm   = (wave & 3) * 32;
  const int wn   = (wave >> 2) * 32;
  const int m0   = blockIdx.y * 128;
  const int n0   = blockIdx.x * 64;

  const unsigned int* Ag  = (const unsigned int*)A;
  const unsigned int* W1g = (const unsigned int*)W1;
  const unsigned int* W2g = (const unsigned int*)W2;
  const int Ku = K >> 1;

  v8f acc1[2][2], acc2[2][2];
#pragma unroll
  for (int t = 0; t < 2; ++t)
#pragma unroll
    for (int j = 0; j < 2; ++j) { acc1[t][j] = zero8(); acc2[t][j] = zero8(); }

  for (int k0 = 0; k0 < K; k0 += 32) {
    const int kc = k0 >> 1;
    __syncthreads();
    stage_tile<128>(Ag,  sA,  m0, Ku, kc, tid);
    stage_tile<64 >(W1g, sB1, n0, Ku, kc, tid);
    stage_tile<64 >(W2g, sB2, n0, Ku, kc, tid);
    tiles_ready();

    v16bf af[2], b1[2], b2[2];
#pragma unroll
    for (int t = 0; t < 2; ++t) af[t] = frag_a(sA, wm + t * 16, lane);
#pragma unroll
    for (int j = 0; j < 2; ++j) { b1[j] = frag_b(sB1, wn + j * 16, lane);
                                  b2[j] = frag_b(sB2, wn + j * 16, lane); }
#pragma unroll
    for (int t = 0; t < 2; ++t)
#pragma unroll
      for (int j = 0; j < 2; ++j) {
        acc1[t][j] = __builtin_amdgcn_wmma_f32_16x16x32_bf16(
            false, af[t], false, b1[j], (short)0, acc1[t][j], false, false);
        acc2[t][j] = __builtin_amdgcn_wmma_f32_16x16x32_bf16(
            false, af[t], false, b2[j], (short)0, acc2[t][j], false, false);
      }
  }

  const int colL  = lane & 15;
  const int rhalf = (lane >> 4) * 8;
#pragma unroll
  for (int t = 0; t < 2; ++t)
#pragma unroll
    for (int j = 0; j < 2; ++j) {
      const int col  = n0 + wn + j * 16 + colL;
      const int rowb = m0 + wm + t * 16 + rhalf;
#pragma unroll
      for (int r = 0; r < 8; ++r) {
        const float g = silu_f(acc1[t][j][r]);
        O[(size_t)(rowb + r) * N + col] = f2bf(g * acc2[t][j][r]);
      }
    }
}

// ---------------------------------------------------------------------------
// Elementwise / reduction kernels
// ---------------------------------------------------------------------------
__global__ void f2bf_kernel(const float* __restrict__ s, unsigned short* __restrict__ d, int n) {
  const int i = blockIdx.x * BDIM + threadIdx.x;
  if (i < n) d[i] = f2bf(s[i]);
}

__global__ void rmsnorm_kernel(const float* __restrict__ x, const float* __restrict__ w,
                               unsigned short* __restrict__ out, int D) {
  __shared__ float red[BDIM];
  const int row = blockIdx.x;
  const float* xr = x + (size_t)row * D;
  float s = 0.f;
  for (int i = threadIdx.x; i < D; i += BDIM) { const float v = xr[i]; s += v * v; }
  red[threadIdx.x] = s;
  __syncthreads();
  for (int off = BDIM / 2; off > 0; off >>= 1) {
    if (threadIdx.x < off) red[threadIdx.x] += red[threadIdx.x + off];
    __syncthreads();
  }
  const float rs = rsqrtf(red[0] / (float)D + 1e-5f);
  unsigned short* orow = out + (size_t)row * D;
  for (int i = threadIdx.x; i < D; i += BDIM) orow[i] = f2bf(xr[i] * rs * w[i]);
}

// depthwise causal conv (D_CONV=4) + SiLU over u = xz[:, :DI]; ld(xz)=2*DI
__global__ void conv_silu_kernel(const float* __restrict__ xz, const float* __restrict__ cw,
                                 const float* __restrict__ cb, float* __restrict__ uaf,
                                 unsigned short* __restrict__ uab) {
  const int idx = blockIdx.x * BDIM + threadIdx.x;     // (b,l,d) flat, d fastest
  const int d   = idx & (DI - 1);
  const int tok = idx >> 11;                           // DI = 2048
  const int l   = tok & (Lc - 1);
  float s = cb[d];
#pragma unroll
  for (int j = 0; j < 4; ++j) {
    const int ls = l - 3 + j;
    if (ls >= 0) s += xz[(size_t)(tok - l + ls) * (2 * DI) + d] * cw[d * 4 + j];
  }
  const float v = silu_f(s);
  uaf[idx] = v;
  uab[idx] = f2bf(v);
}

__global__ void slice_dtr_kernel(const float* __restrict__ proj, unsigned short* __restrict__ dtr) {
  const int i = blockIdx.x * BDIM + threadIdx.x;       // Mtok * DTR
  const int row = i >> 6, c = i & 63;
  dtr[i] = f2bf(proj[row * 96 + c]);
}

// selective scan: one thread per (b,d); 16 states in registers; B/C staged in LDS
__global__ void scan_kernel(const float* __restrict__ dt, const float* __restrict__ ua,
                            const float* __restrict__ proj, const float* __restrict__ A_log,
                            const float* __restrict__ Dp, float* __restrict__ y) {
  __shared__ float sBC[64][32];
  const int gid = blockIdx.x * BDIM + threadIdx.x;     // b constant per block
  const int b = gid >> 11;
  const int d = gid & (DI - 1);
  float Ac[16], h[16];
#pragma unroll
  for (int n = 0; n < 16; ++n) { Ac[n] = -__expf(A_log[d * 16 + n]); h[n] = 0.f; }
  const float Dd = Dp[d];
  for (int l0 = 0; l0 < Lc; l0 += 64) {
    __syncthreads();
    for (int i = threadIdx.x; i < 64 * 32; i += BDIM) {
      const int li = i >> 5, j = i & 31;
      sBC[li][j] = proj[(size_t)(b * Lc + l0 + li) * 96 + 64 + j];
    }
    __syncthreads();
    for (int li = 0; li < 64; ++li) {
      const size_t base = (size_t)(b * Lc + l0 + li) * DI + d;
      const float dtv = dt[base];
      const float uv  = ua[base];
      float acc = 0.f;
#pragma unroll
      for (int n = 0; n < 16; ++n) {
        const float dA = __expf(dtv * Ac[n]);
        h[n] = dA * h[n] + dtv * sBC[li][n] * uv;
        acc += h[n] * sBC[li][16 + n];
      }
      y[base] = acc + uv * Dd;
    }
  }
}

// y_gated = bf16( y * silu(z) ), z = xz[:, DI:]
__global__ void gate_kernel(const float* __restrict__ y, const float* __restrict__ xz,
                            unsigned short* __restrict__ out) {
  const int i = blockIdx.x * BDIM + threadIdx.x;
  const int d = i & (DI - 1);
  const int tok = i >> 11;
  const float z = xz[(size_t)tok * (2 * DI) + DI + d];
  out[i] = f2bf(y[i] * silu_f(z));
}

// ---------------------------------------------------------------------------
extern "C" void kernel_launch(void* const* d_in, const int* in_sizes, int n_in,
                              void* d_out, int out_size, void* d_ws, size_t ws_size,
                              hipStream_t stream) {
  (void)in_sizes; (void)n_in; (void)out_size; (void)ws_size;
  const float* x          = (const float*)d_in[0];
  const float* n1w        = (const float*)d_in[1];
  const float* n2w        = (const float*)d_in[2];
  const float* in_proj_w  = (const float*)d_in[3];
  const float* conv_w     = (const float*)d_in[4];
  const float* conv_b     = (const float*)d_in[5];
  const float* x_proj_w   = (const float*)d_in[6];
  const float* dt_proj_w  = (const float*)d_in[7];
  const float* dt_proj_b  = (const float*)d_in[8];
  const float* A_log      = (const float*)d_in[9];
  const float* Dvec       = (const float*)d_in[10];
  const float* out_proj_w = (const float*)d_in[11];
  const float* w1         = (const float*)d_in[12];
  const float* w2         = (const float*)d_in[13];
  const float* w3         = (const float*)d_in[14];
  float* out = (float*)d_out;

  char* base = (char*)d_ws;
  size_t off = 0;
  auto alloc = [&](size_t bytes) -> void* {
    off = (off + 255) & ~(size_t)255;
    void* r = base + off;
    off += bytes;
    return r;
  };

  // bf16 weight copies
  unsigned short* w_inp = (unsigned short*)alloc((size_t)2 * DI * DM * 2);
  unsigned short* w_xp  = (unsigned short*)alloc((size_t)(DTR + 2 * DSt) * DI * 2);
  unsigned short* w_dt  = (unsigned short*)alloc((size_t)DI * DTR * 2);
  unsigned short* w_op  = (unsigned short*)alloc((size_t)DM * DI * 2);
  unsigned short* w1b   = (unsigned short*)alloc((size_t)DFF * DM * 2);
  unsigned short* w2b   = (unsigned short*)alloc((size_t)DFF * DM * 2);
  unsigned short* w3b   = (unsigned short*)alloc((size_t)DM * DFF * 2);
  // activations
  unsigned short* xn_bf   = (unsigned short*)alloc((size_t)Mtok * DM * 2);
  float*          xz      = (float*)alloc((size_t)Mtok * 2 * DI * 4);
  float*          uaf     = (float*)alloc((size_t)Mtok * DI * 4);
  unsigned short* uab     = (unsigned short*)alloc((size_t)Mtok * DI * 2);
  float*          proj    = (float*)alloc((size_t)Mtok * 96 * 4);
  unsigned short* dtr_bf  = (unsigned short*)alloc((size_t)Mtok * DTR * 2);
  float*          dtb     = (float*)alloc((size_t)Mtok * DI * 4);
  float*          ybuf    = (float*)alloc((size_t)Mtok * DI * 4);
  unsigned short* yg_bf   = (unsigned short*)alloc((size_t)Mtok * DI * 2);
  float*          hres    = (float*)alloc((size_t)Mtok * DM * 4);
  unsigned short* xn2_bf  = (unsigned short*)alloc((size_t)Mtok * DM * 2);
  unsigned short* hmid_bf = (unsigned short*)alloc((size_t)Mtok * DFF * 2);

  auto cvt = [&](const float* s, unsigned short* d, int n) {
    f2bf_kernel<<<(n + BDIM - 1) / BDIM, BDIM, 0, stream>>>(s, d, n);
  };
  cvt(in_proj_w,  w_inp, 2 * DI * DM);
  cvt(x_proj_w,   w_xp,  (DTR + 2 * DSt) * DI);
  cvt(dt_proj_w,  w_dt,  DI * DTR);
  cvt(out_proj_w, w_op,  DM * DI);
  cvt(w1, w1b, DFF * DM);
  cvt(w2, w2b, DFF * DM);
  cvt(w3, w3b, DM * DFF);

  // 1) RMSNorm1 -> bf16
  rmsnorm_kernel<<<Mtok, BDIM, 0, stream>>>(x, n1w, xn_bf, DM);
  // 2) in_proj: xz = xn @ in_proj_w^T
  gemm_nt<0><<<dim3((2 * DI) / 128, Mtok / 128), BDIM, 0, stream>>>(
      xn_bf, w_inp, xz, nullptr, nullptr, Mtok, 2 * DI, DM);
  // 3) causal depthwise conv + SiLU on u
  conv_silu_kernel<<<(Mtok * DI) / BDIM, BDIM, 0, stream>>>(xz, conv_w, conv_b, uaf, uab);
  // 4) x_proj: proj = u_act @ x_proj_w^T  (N=96, guarded)
  gemm_nt<0><<<dim3(1, Mtok / 128), BDIM, 0, stream>>>(
      uab, w_xp, proj, nullptr, nullptr, Mtok, 96, DI);
  // 5) slice dt_r -> bf16
  slice_dtr_kernel<<<(Mtok * DTR) / BDIM, BDIM, 0, stream>>>(proj, dtr_bf);
  // 6) dt = softplus(dt_r @ dt_proj_w^T + b)
  gemm_nt<1><<<dim3(DI / 128, Mtok / 128), BDIM, 0, stream>>>(
      dtr_bf, w_dt, dtb, dt_proj_b, nullptr, Mtok, DI, DTR);
  // 7) selective scan -> y
  scan_kernel<<<(Bc * DI) / BDIM, BDIM, 0, stream>>>(dtb, uaf, proj, A_log, Dvec, ybuf);
  // 8) gate: y * silu(z) -> bf16
  gate_kernel<<<(Mtok * DI) / BDIM, BDIM, 0, stream>>>(ybuf, xz, yg_bf);
  // 9) out_proj + residual x -> h
  gemm_nt<2><<<dim3(DM / 128, Mtok / 128), BDIM, 0, stream>>>(
      yg_bf, w_op, hres, nullptr, x, Mtok, DM, DI);
  // 10) RMSNorm2 -> bf16
  rmsnorm_kernel<<<Mtok, BDIM, 0, stream>>>(hres, n2w, xn2_bf, DM);
  // 11) fused SwiGLU dual GEMM -> bf16 hidden
  gemm_dual_swiglu<<<dim3(DFF / 64, Mtok / 128), BDIM, 0, stream>>>(
      xn2_bf, w1b, w2b, hmid_bf, Mtok, DFF, DM);
  // 12) w3 GEMM + residual h -> out
  gemm_nt<2><<<dim3(DM / 128, Mtok / 128), BDIM, 0, stream>>>(
      hmid_bf, w3b, out, nullptr, hres, Mtok, DM, DFF);
}
